// ErrorNeurons_644245094709
// MI455X (gfx1250) — compile-verified, matrix operand order checked by reading, not run on hardware
//
#include <hip/hip_runtime.h>

typedef __attribute__((ext_vector_type(16))) _Float16 v16h;
typedef __attribute__((ext_vector_type(8)))  _Float16 v8h;
typedef __attribute__((ext_vector_type(8)))  float    v8f;

#define CSZ     5
#define HGT     128
#define WID     128
#define NBAT    4
#define HW      (HGT*WID)
#define DPAD    32              // 25-dim descriptor padded to K=32
#define LN_EPS  1e-5f
#define COS_EPS 1e-8f

__device__ __forceinline__ int imin(int a, int b) { return a < b ? a : b; }
__device__ __forceinline__ int imax(int a, int b) { return a > b ? a : b; }

// ---------------------------------------------------------------------------
// Stage 1: per-pixel 5x5 LayerNorm (+2.0) -> FP16 descriptor padded to 32,
// plus f32 inverse L2 norm. One thread per pixel.
// ---------------------------------------------------------------------------
__global__ __launch_bounds__(256) void bu_kernel(const float* __restrict__ img,
                                                 _Float16* __restrict__ bu,
                                                 float* __restrict__ invn)
{
    int id  = blockIdx.x * 256 + threadIdx.x;   // 0 .. NBAT*HW-1
    int b   = id >> 14;
    int rem = id & (HW - 1);
    int y   = rem >> 7;
    int x   = rem & (WID - 1);

    const float* im = img + (size_t)b * HW;

    float v[25];
    float sum = 0.f;
#pragma unroll
    for (int i = 0; i < CSZ; ++i) {
#pragma unroll
        for (int j = 0; j < CSZ; ++j) {
            int yy = y + i - 2, xx = x + j - 2;
            float t = (yy >= 0 && yy < HGT && xx >= 0 && xx < WID)
                          ? im[yy * WID + xx] : 0.f;
            v[i * CSZ + j] = t;
            sum += t;
        }
    }
    float m = sum * (1.f / 25.f);
    float var = 0.f;
#pragma unroll
    for (int k = 0; k < 25; ++k) { float d = v[k] - m; var += d * d; }
    var *= (1.f / 25.f);
    float inv = rsqrtf(var + LN_EPS);

    _Float16* o = bu + (size_t)id * DPAD;
    float nsq = 0.f;
#pragma unroll
    for (int k = 0; k < 25; ++k) {
        float t = (v[k] - m) * inv + 2.f;
        nsq += t * t;
        o[k] = (_Float16)t;
    }
#pragma unroll
    for (int k = 25; k < DPAD; ++k) o[k] = (_Float16)0.f;  // ws is poisoned; rewrite pad

    invn[id] = 1.f / fmaxf(sqrtf(nsq), COS_EPS);
}

// ---------------------------------------------------------------------------
// Stage 2: one wave32 per (batch, row, 16-pixel x-tile).
// A = 16 row pixels (16x32 f16). For each dy in [-4,4]: two B tiles
// (columns x0-4..x0+11 and x0+12..x0+27) -> 2x v_wmma_f32_16x16x32_f16.
// Inner loop is maskless (16 FMAs); the dy-independent band/bounds weights
// (plus center-exclusion for dy==0) are applied once after the loop.
// ---------------------------------------------------------------------------
__global__ __launch_bounds__(256) void sal_kernel(const _Float16* __restrict__ bu,
                                                  const float* __restrict__ invn,
                                                  float* __restrict__ out)
{
    const int lane = threadIdx.x & 31;
    const int wave = blockIdx.x * (256 / 32) + (threadIdx.x >> 5); // 0..4095
    const int xt = wave & 7;                // WID/16 = 8 tiles
    const int y  = (wave >> 3) & (HGT - 1);
    const int b  = wave >> 10;
    const int x0 = xt * 16;

    const int n = lane & 15;                // matrix row (A) / column (B) index
    const int g = lane >> 4;                // K-chunk selector per ISA layout

    const _Float16* base = bu + (size_t)b * HW * DPAD;
    const float*    inb  = invn + (size_t)b * HW;

    union V16 { v16h v; v8h h[2]; };

    // A: rows m=0..15 -> pixels (y, x0+m); this lane supplies row m=n.
    V16 A;
    {
        const char* ar = (const char*)(base + (size_t)(y * WID + x0 + n) * DPAD);
        A.h[0] = *(const v8h*)(ar + g * 16);
        A.h[1] = *(const v8h*)(ar + 32 + g * 16);
    }

    // B-tile column coordinates (dy-independent).
    const int xn0 = x0 - 4 + n;             // tile 0 global x
    const int xn1 = x0 + 12 + n;            // tile 1 global x
    const int xc0 = imin(imax(xn0, 0), WID - 1);
    const int xc1 = imin(imax(xn1, 0), WID - 1);
    const bool xok0 = (xn0 >= 0) && (xn0 < WID);
    const bool xok1 = (xn1 < WID);

    // Per-(lane, r) 0/1 weights: w* for dy!=0 rows, w*z additionally drops
    // the center (dx==0) for the dy==0 row.
    float w0[8], w1[8], w0z[8], w1z[8];
#pragma unroll
    for (int r = 0; r < 8; ++r) {
        int mrow = r + 8 * g;
        int dx0 = xn0 - (x0 + mrow);
        int dx1 = xn1 - (x0 + mrow);
        bool in0 = xok0 && (dx0 >= -4) && (dx0 <= 4);
        bool in1 = xok1 && (dx1 >= -4) && (dx1 <= 4);
        w0[r]  = in0 ? 1.f : 0.f;
        w1[r]  = in1 ? 1.f : 0.f;
        w0z[r] = (in0 && dx0 != 0) ? 1.f : 0.f;
        w1z[r] = (in1 && dx1 != 0) ? 1.f : 0.f;
    }

    float t0[8], t1[8], e0[8], e1[8];
#pragma unroll
    for (int r = 0; r < 8; ++r) { t0[r] = 0.f; t1[r] = 0.f; e0[r] = 0.f; e1[r] = 0.f; }

    for (int dy = -4; dy <= 4; ++dy) {
        int yy = y + dy;
        if (yy < 0 || yy >= HGT) continue;  // wave-uniform: EXEC stays all-ones

        const char* rb = (const char*)(base + (size_t)yy * WID * DPAD);
        V16 B0, B1;
        B0.h[0] = *(const v8h*)(rb + xc0 * (DPAD * 2) + g * 16);
        B0.h[1] = *(const v8h*)(rb + xc0 * (DPAD * 2) + 32 + g * 16);
        B1.h[0] = *(const v8h*)(rb + xc1 * (DPAD * 2) + g * 16);
        B1.h[1] = *(const v8h*)(rb + xc1 * (DPAD * 2) + 32 + g * 16);

        v8f d0 = {};
        v8f d1 = {};
        d0 = __builtin_amdgcn_wmma_f32_16x16x32_f16(false, A.v, false, B0.v,
                                                    (short)0, d0, false, false);
        d1 = __builtin_amdgcn_wmma_f32_16x16x32_f16(false, A.v, false, B1.v,
                                                    (short)0, d1, false, false);

        float in0 = inb[yy * WID + xc0];    // neighbour inverse norms
        float in1 = inb[yy * WID + xc1];

        if (dy != 0) {                      // uniform branch
#pragma unroll
            for (int r = 0; r < 8; ++r) {
                t0[r] = fmaf(d0[r], in0, t0[r]);
                t1[r] = fmaf(d1[r], in1, t1[r]);
            }
        } else {
#pragma unroll
            for (int r = 0; r < 8; ++r) {
                e0[r] = d0[r] * in0;
                e1[r] = d1[r] * in1;
            }
        }
    }

    // Apply dy-independent weights once, then reduce across each 16-lane half.
    float s[8];
#pragma unroll
    for (int r = 0; r < 8; ++r) {
        s[r] = fmaf(w0[r], t0[r],
               fmaf(w1[r], t1[r],
               fmaf(w0z[r], e0[r], w1z[r] * e1[r])));
        s[r] += __shfl_xor(s[r], 1, 32);
        s[r] += __shfl_xor(s[r], 2, 32);
        s[r] += __shfl_xor(s[r], 4, 32);
        s[r] += __shfl_xor(s[r], 8, 32);
    }

    int rsel = lane & 15;
    if (rsel < 8) {
        float acc = 0.f;
#pragma unroll
        for (int r = 0; r < 8; ++r) acc = (rsel == r) ? s[r] : acc;

        int mrow = rsel + 8 * g;
        int x = x0 + mrow;
        float invm = inb[y * WID + x];
        int ndx = imin(x, 4) + imin(WID - 1 - x, 4) + 1;
        int ndy = imin(y, 4) + imin(HGT - 1 - y, 4) + 1;
        float cnt = (float)(ndx * ndy - 1);
        // sal = mean((1-cos)/2) = 0.5 - 0.5 * (sum cos)/cnt
        out[(size_t)b * HW + y * WID + x] = 0.5f - 0.5f * acc * invm / cnt;
    }
}

extern "C" void kernel_launch(void* const* d_in, const int* in_sizes, int n_in,
                              void* d_out, int out_size, void* d_ws, size_t ws_size,
                              hipStream_t stream) {
    (void)in_sizes; (void)n_in; (void)out_size; (void)ws_size;

    const float* img = (const float*)d_in[0];
    float* out = (float*)d_out;

    _Float16* bu = (_Float16*)d_ws;                                       // 4 MiB
    float* invn  = (float*)((char*)d_ws +
                            (size_t)NBAT * HW * DPAD * sizeof(_Float16)); // +256 KiB

    // Stage 1: 65536 pixels, one thread each.
    bu_kernel<<<(NBAT * HW) / 256, 256, 0, stream>>>(img, bu, invn);

    // Stage 2: 4096 waves (one per 16-pixel row tile) = 512 blocks x 8 waves.
    sal_kernel<<<(NBAT * HGT * (WID / 16) * 32) / 256, 256, 0, stream>>>(bu, invn, out);
}
